// CasualSelfAttention_71751723647208
// MI455X (gfx1250) — compile-verified, matrix operand order checked by reading, not run on hardware
//
#include <hip/hip_runtime.h>
#include <math.h>

// ---------------------------------------------------------------------------
// CDNA5 (gfx1250) causal self-attention: WMMA bf16 + async global->LDS staging.
// ---------------------------------------------------------------------------

typedef __attribute__((ext_vector_type(16))) __bf16 v16bf;
typedef __attribute__((ext_vector_type(8)))  float  v8f;

union Frag16 {
    uint4 u[2];   // 32 bytes = 16 bf16
    v16bf v;
};

__device__ __forceinline__ unsigned short f2bf(float f) {
    unsigned int u = __float_as_uint(f);
    unsigned int r = u + 0x7FFFu + ((u >> 16) & 1u);   // round-to-nearest-even
    return (unsigned short)(r >> 16);
}
__device__ __forceinline__ float bf2f(unsigned short h) {
    return __uint_as_float(((unsigned int)h) << 16);
}

// Issue one 16-byte async copy global -> LDS (tracked by ASYNCcnt).
__device__ __forceinline__ void async_copy_b128(unsigned int lds_addr,
                                                const void* gaddr) {
    asm volatile("global_load_async_to_lds_b128 %0, %1, off"
                 :: "v"(lds_addr), "v"(gaddr) : "memory");
}
__device__ __forceinline__ void wait_async0() {
    asm volatile("s_wait_asynccnt 0x0" ::: "memory");
}

// Problem constants (match setup_inputs)
#define BB 2
#define TT 2048
#define CC 1024
#define HH 16
#define DD 64

// ---------------------------------------------------------------------------
// Elementwise fp32 -> bf16
// ---------------------------------------------------------------------------
__global__ void k_cvt_bf16(const float* __restrict__ in,
                           unsigned short* __restrict__ out, int n) {
    int i = blockIdx.x * blockDim.x + threadIdx.x;
    if (i < n) out[i] = f2bf(in[i]);
}

// fp32 [K,N] -> bf16 transposed [N,K]
__global__ void k_cvt_bf16_T(const float* __restrict__ in,
                             unsigned short* __restrict__ out, int K, int N) {
    int i = blockIdx.x * blockDim.x + threadIdx.x;
    if (i < K * N) {
        int k = i / N, n = i % N;
        out[(size_t)n * K + k] = f2bf(in[i]);
    }
}

// ---------------------------------------------------------------------------
// WMMA GEMM: out[M,N] = A[M,K](bf16) * Bt[N,K](bf16)^T + bias
// One wave computes a 16x64 tile: A fragment reused across 4 B fragments.
// ---------------------------------------------------------------------------
__global__ void k_gemm_wmma_bf16(const unsigned short* __restrict__ A,
                                 const unsigned short* __restrict__ Bt,
                                 const float* __restrict__ bias,
                                 unsigned short* __restrict__ outb,
                                 float* __restrict__ outf,
                                 int M, int N, int K) {
    const int nb   = blockIdx.x * 64;
    const int mb   = blockIdx.y * 16;
    const int lane = threadIdx.x;
    const int half = lane >> 4;   // 0: k 0-7/16-23, 1: k 8-15/24-31
    const int l16  = lane & 15;

    const unsigned short* Ar = A  + (size_t)(mb + l16) * K;
    const unsigned short* Br = Bt + (size_t)(nb + l16) * K;

    v8f acc[4];
#pragma unroll
    for (int j = 0; j < 4; ++j)
#pragma unroll
        for (int r = 0; r < 8; ++r) acc[j][r] = 0.0f;

    for (int kc = 0; kc < K; kc += 32) {
        const int s = kc + half * 8;
        Frag16 a;
        a.u[0] = *(const uint4*)(Ar + s);
        a.u[1] = *(const uint4*)(Ar + s + 16);
        __builtin_prefetch(Ar + s + 32, 0, 0);
#pragma unroll
        for (int j = 0; j < 4; ++j) {
            const unsigned short* Bj = Br + (size_t)j * 16 * K + s;
            Frag16 b;
            b.u[0] = *(const uint4*)(Bj);
            b.u[1] = *(const uint4*)(Bj + 16);
            acc[j] = __builtin_amdgcn_wmma_f32_16x16x32_bf16(
                false, a.v, false, b.v, (short)0, acc[j], false, false);
        }
    }

#pragma unroll
    for (int j = 0; j < 4; ++j) {
        const int n = nb + j * 16 + l16;
        const float bv = bias[n];
#pragma unroll
        for (int r = 0; r < 8; ++r) {
            const int m = mb + r + half * 8;
            const float val = acc[j][r] + bv;
            if (outf) outf[(size_t)m * N + n] = val;
            else      outb[(size_t)m * N + n] = f2bf(val);
        }
    }
}

// ---------------------------------------------------------------------------
// RoPE + head split: qkv bf16 [B,T,3C] ->
//   qh,kh bf16 [B,H,T,D] (roped), vt bf16 [B,H,D,T] (transposed)
// ---------------------------------------------------------------------------
__global__ void k_rope_split(const unsigned short* __restrict__ qkv,
                             unsigned short* __restrict__ qh,
                             unsigned short* __restrict__ kh,
                             unsigned short* __restrict__ vt) {
    const int idx = blockIdx.x * blockDim.x + threadIdx.x;
    const int total = BB * HH * TT * (DD / 2);
    if (idx >= total) return;
    const int i = idx & 31;
    const int t = (idx >> 5) & (TT - 1);
    const int h = (idx >> 5) / TT % HH;
    const int b = (idx >> 5) / TT / HH;

    const size_t base = ((size_t)b * TT + t) * (3 * CC) + h * DD;
    const float q1 = bf2f(qkv[base + i]);
    const float q2 = bf2f(qkv[base + i + 32]);
    const float k1 = bf2f(qkv[base + CC + i]);
    const float k2 = bf2f(qkv[base + CC + i + 32]);
    const float v1 = bf2f(qkv[base + 2 * CC + i]);
    const float v2 = bf2f(qkv[base + 2 * CC + i + 32]);

    const float freq = __powf(10000.0f, -2.0f * (float)i / 64.0f);
    float s, c;
    __sincosf((float)t * freq, &s, &c);

    const size_t hb = (((size_t)b * HH + h) * TT + t) * DD;
    qh[hb + i]      = f2bf(q1 * c - q2 * s);
    qh[hb + i + 32] = f2bf(q1 * s + q2 * c);
    kh[hb + i]      = f2bf(k1 * c - k2 * s);
    kh[hb + i + 32] = f2bf(k1 * s + k2 * c);

    const size_t vb = ((size_t)b * HH + h) * DD;
    vt[(vb + i)      * TT + t] = f2bf(v1);
    vt[(vb + i + 32) * TT + t] = f2bf(v2);
}

// ---------------------------------------------------------------------------
// Flash attention: 4 waves/block = 4 adjacent 16-row query tiles.
// Each 32-key K/V chunk is staged once into LDS with async global->LDS copies
// (ASYNCcnt), then all 4 waves consume it via ds_load_b128 fragments.
// ---------------------------------------------------------------------------
__global__ void __launch_bounds__(128)
k_flash_attn(const unsigned short* __restrict__ qh,
             const unsigned short* __restrict__ kh,
             const unsigned short* __restrict__ vt,
             unsigned short* __restrict__ yh) {
    const int bh   = blockIdx.y;          // 0..B*H-1
    const int h    = bh % HH;
    const int b    = bh / HH;
    const int tid  = threadIdx.x;         // 0..127
    const int wave = tid >> 5;            // 0..3
    const int lane = tid & 31;
    const int half = lane >> 4;
    const int l16  = lane & 15;
    const int qbase = blockIdx.x * 64 + wave * 16;

    __shared__ __align__(16) unsigned short Klds[32 * 64];      // [key][d]
    __shared__ __align__(16) unsigned short Vlds[64 * 32];      // [d][key]
    __shared__ __align__(16) unsigned short Plds[4][16 * 32];   // per-wave P

    const unsigned short* Kb  = kh + (size_t)bh * TT * DD;
    const unsigned short* Vtb = vt + (size_t)bh * DD * TT;

    // Q fragments (d 0..31 and 32..63), loaded once per wave
    Frag16 q0, q1;
    {
        const unsigned short* Qr =
            qh + ((size_t)bh * TT + qbase + l16) * DD + half * 8;
        q0.u[0] = *(const uint4*)(Qr);
        q0.u[1] = *(const uint4*)(Qr + 16);
        q1.u[0] = *(const uint4*)(Qr + 32);
        q1.u[1] = *(const uint4*)(Qr + 48);
    }

    float mrow[8], lrow[8];
    v8f oacc[4];
#pragma unroll
    for (int r = 0; r < 8; ++r) { mrow[r] = -3.0e38f; lrow[r] = 0.0f; }
#pragma unroll
    for (int nt = 0; nt < 4; ++nt)
#pragma unroll
        for (int r = 0; r < 8; ++r) oacc[nt][r] = 0.0f;

    const float scale = 0.125f;   // 1/sqrt(64)
    const unsigned int kbase_lds = (unsigned int)(uintptr_t)&Klds[0];
    const unsigned int vbase_lds = (unsigned int)(uintptr_t)&Vlds[0];
    const int kc_end = blockIdx.x * 64 + 64;   // uniform across the block

    for (int kc = 0; kc < kc_end; kc += 32) {
        // ---- stage K (4KB, contiguous) and V (64 x 64B rows) into LDS ----
        const char* Ksrc = (const char*)(Kb + (size_t)kc * DD);
#pragma unroll
        for (int jj = 0; jj < 2; ++jj) {
            const int j = tid + jj * 128;             // 0..255 16B blocks
            async_copy_b128(kbase_lds + j * 16, Ksrc + j * 16);
            const int dim = j >> 2, seg = j & 3;
            const char* Vsrc =
                (const char*)(Vtb + (size_t)dim * TT + kc) + seg * 16;
            async_copy_b128(vbase_lds + j * 16, Vsrc);
        }
        wait_async0();
        __syncthreads();                              // K/V visible to all waves

        // ---- S = Q K^T for 32 keys (fragments from LDS) ----
        v8f s0 = {}, s1 = {};
#pragma unroll
        for (int f = 0; f < 2; ++f) {
            const unsigned short* Kr = &Klds[(f * 16 + l16) * 64 + half * 8];
            Frag16 b0, b1;
            b0.u[0] = *(const uint4*)(Kr);
            b0.u[1] = *(const uint4*)(Kr + 16);
            b1.u[0] = *(const uint4*)(Kr + 32);
            b1.u[1] = *(const uint4*)(Kr + 48);
            if (f == 0) {
                s0 = __builtin_amdgcn_wmma_f32_16x16x32_bf16(
                    false, q0.v, false, b0.v, (short)0, s0, false, false);
                s0 = __builtin_amdgcn_wmma_f32_16x16x32_bf16(
                    false, q1.v, false, b1.v, (short)0, s0, false, false);
            } else {
                s1 = __builtin_amdgcn_wmma_f32_16x16x32_bf16(
                    false, q0.v, false, b0.v, (short)0, s1, false, false);
                s1 = __builtin_amdgcn_wmma_f32_16x16x32_bf16(
                    false, q1.v, false, b1.v, (short)0, s1, false, false);
            }
        }

        // ---- causal mask + online softmax ----
#pragma unroll
        for (int r = 0; r < 8; ++r) {
            const int m    = r + half * 8;
            const int qidx = qbase + m;
            float e0 = (kc + l16      <= qidx) ? s0[r] * scale : -3.0e38f;
            float e1 = (kc + 16 + l16 <= qidx) ? s1[r] * scale : -3.0e38f;

            float mx = fmaxf(e0, e1);
            mx = fmaxf(mx, __shfl_xor(mx, 1, 32));
            mx = fmaxf(mx, __shfl_xor(mx, 2, 32));
            mx = fmaxf(mx, __shfl_xor(mx, 4, 32));
            mx = fmaxf(mx, __shfl_xor(mx, 8, 32));

            const float mnew = fmaxf(mrow[r], mx);
            const float corr = __expf(mrow[r] - mnew);
            mrow[r] = mnew;

            const float p0 = __expf(e0 - mnew);
            const float p1 = __expf(e1 - mnew);
            float sum = p0 + p1;
            sum += __shfl_xor(sum, 1, 32);
            sum += __shfl_xor(sum, 2, 32);
            sum += __shfl_xor(sum, 4, 32);
            sum += __shfl_xor(sum, 8, 32);
            lrow[r] = lrow[r] * corr + sum;

            oacc[0][r] *= corr; oacc[1][r] *= corr;
            oacc[2][r] *= corr; oacc[3][r] *= corr;

            Plds[wave][m * 32 + l16]      = f2bf(p0);
            Plds[wave][m * 32 + 16 + l16] = f2bf(p1);
        }
        __syncthreads();                              // P visible; S-phase done

        // ---- P (C-frag) -> A-frag via LDS, then O += P V ----
        Frag16 pa;
        pa.u[0] = *(const uint4*)(&Plds[wave][l16 * 32 + half * 8]);
        pa.u[1] = *(const uint4*)(&Plds[wave][l16 * 32 + half * 8 + 16]);
#pragma unroll
        for (int nt = 0; nt < 4; ++nt) {
            const unsigned short* Vr = &Vlds[(nt * 16 + l16) * 32 + half * 8];
            Frag16 bv;
            bv.u[0] = *(const uint4*)(Vr);
            bv.u[1] = *(const uint4*)(Vr + 16);
            oacc[nt] = __builtin_amdgcn_wmma_f32_16x16x32_bf16(
                false, pa.v, false, bv.v, (short)0, oacc[nt], false, false);
        }
        __syncthreads();                              // done reading K/V/P
    }

    // ---- epilogue: normalize and store y [B,T,C] bf16 ----
    const size_t ybase = ((size_t)b * TT + qbase) * CC + h * DD;
#pragma unroll
    for (int r = 0; r < 8; ++r) {
        const int m = r + half * 8;
        const float inv = 1.0f / lrow[r];
#pragma unroll
        for (int nt = 0; nt < 4; ++nt)
            yh[ybase + (size_t)m * CC + nt * 16 + l16] = f2bf(oacc[nt][r] * inv);
    }
}

// ---------------------------------------------------------------------------
// Host launcher
// ---------------------------------------------------------------------------
extern "C" void kernel_launch(void* const* d_in, const int* in_sizes, int n_in,
                              void* d_out, int out_size, void* d_ws, size_t ws_size,
                              hipStream_t stream) {
    const float* x      = (const float*)d_in[0];
    const float* w_attn = (const float*)d_in[1];
    const float* b_attn = (const float*)d_in[2];
    const float* w_proj = (const float*)d_in[3];
    const float* b_proj = (const float*)d_in[4];
    float* out = (float*)d_out;

    const int M  = BB * TT;          // 4096
    const int K  = CC;               // 1024
    const int N3 = 3 * CC;           // 3072

    char* ws = (char*)d_ws;
    size_t off = 0;
    unsigned short* xb   = (unsigned short*)(ws + off); off += (size_t)M * CC * 2;
    unsigned short* wab  = (unsigned short*)(ws + off); off += (size_t)N3 * CC * 2;
    unsigned short* wpb  = (unsigned short*)(ws + off); off += (size_t)CC * CC * 2;
    unsigned short* qkvb = (unsigned short*)(ws + off); off += (size_t)M * N3 * 2;
    unsigned short* qh   = (unsigned short*)(ws + off); off += (size_t)M * CC * 2;
    unsigned short* kh   = (unsigned short*)(ws + off); off += (size_t)M * CC * 2;
    unsigned short* vt   = (unsigned short*)(ws + off); off += (size_t)M * CC * 2;
    unsigned short* yh   = (unsigned short*)(ws + off); off += (size_t)M * CC * 2;
    (void)ws_size; (void)in_sizes; (void)n_in; (void)out_size;

    // 1) conversions
    {
        int n = M * CC;
        k_cvt_bf16<<<dim3((n + 255) / 256), dim3(256), 0, stream>>>(x, xb, n);
    }
    {
        int n = K * N3;
        k_cvt_bf16_T<<<dim3((n + 255) / 256), dim3(256), 0, stream>>>(w_attn, wab, K, N3);
    }
    {
        int n = K * CC;
        k_cvt_bf16_T<<<dim3((n + 255) / 256), dim3(256), 0, stream>>>(w_proj, wpb, K, CC);
    }

    // 2) qkv = x @ w_attn + b_attn   (bf16 out)
    k_gemm_wmma_bf16<<<dim3(N3 / 64, M / 16), dim3(32), 0, stream>>>(
        xb, wab, b_attn, qkvb, (float*)nullptr, M, N3, K);

    // 3) RoPE + head split
    {
        int n = BB * HH * TT * (DD / 2);
        k_rope_split<<<dim3((n + 255) / 256), dim3(256), 0, stream>>>(qkvb, qh, kh, vt);
    }

    // 4) flash attention (4 query tiles per block share K/V via async LDS)
    k_flash_attn<<<dim3(TT / 64, BB * HH), dim3(128), 0, stream>>>(qh, kh, vt, yh);

    // 5) out = y @ w_proj + b_proj   (fp32 out)
    k_gemm_wmma_bf16<<<dim3(CC / 64, M / 16), dim3(32), 0, stream>>>(
        yh, wpb, b_proj, (unsigned short*)nullptr, out, M, CC, K);
}
